// GNN_13657996002095
// MI455X (gfx1250) — matrix operand order repaired and checked by reading, not compiled, not run on hardware
//
#include <hip/hip_runtime.h>
#include <hip/hip_bf16.h>

// ---------------------------------------------------------------------------
// Types for WMMA (CDNA5 / gfx1250, wave32)
// ---------------------------------------------------------------------------
typedef __attribute__((ext_vector_type(16))) __bf16 v16bf;
typedef __attribute__((ext_vector_type(8)))  __bf16 v8bf;
typedef __attribute__((ext_vector_type(8)))  float  v8f;

#define B_DIM 2
#define N_NODES 5000
#define E_EDGES 160000
#define H_DIM 128

#define WLDS_ELEMS 49152               // 96 KB weight stage (max K*N = 384*128)
#define CLDS_ELEMS 16384               // 32 KB C re-tile stage (8 waves * 2048)
#define LDS_BYTES  ((WLDS_ELEMS + CLDS_ELEMS) * 2)
#define MAX_BLOCKS 640

// ---------------------------------------------------------------------------
// Weight packing: fp32 [srcK, srcN] row-major -> bf16 packed in a contiguous-k
// V_WMMA_*_16X16X32_BF16 B-matrix lane layout (k = 16*(lane/16) + i; A uses the same order):
//   dst[((kt*ntiles + nt)*32 + lane)*16 + i] = W[kt*32 + (lane&16?16:0) + i][nt*16 + (lane&15)]
// Zero-pads K up to padK and N up to padN.
// ---------------------------------------------------------------------------
__global__ void pack_w_kernel(const float* __restrict__ W, __bf16* __restrict__ dst,
                              int srcK, int srcN, int padK, int padN) {
    int t = blockIdx.x * blockDim.x + threadIdx.x;
    int lane = t & 31;
    int g = t >> 5;
    int ntiles = padN >> 4;
    int ktiles = padK >> 5;
    if (g >= ntiles * ktiles) return;
    int nt = g % ntiles;
    int kt = g / ntiles;
    int n  = nt * 16 + (lane & 15);
    int kb = kt * 32 + ((lane & 16) ? 16 : 0);
    v16bf v;
#pragma unroll
    for (int i = 0; i < 16; ++i) {
        int k = kb + i;
        float f = (k < srcK && n < srcN) ? W[(size_t)k * srcN + n] : 0.f;
        v[i] = (__bf16)f;
    }
    *(v16bf*)(dst + ((size_t)g * 32 + lane) * 16) = v;
}

// ---------------------------------------------------------------------------
// A-fragment chunk loader: returns 8 bf16 for columns [c, c+8) of one row.
//  amode 0: plain bf16 matrix (pa = row base)
//  amode 2: fused concat gather [ x[send] | x[recv] | m_prev ]
// Chunks are 16-byte aligned and never straddle the 128-column boundaries.
// ---------------------------------------------------------------------------
__device__ __forceinline__ v8bf loadA8(int amode, const __bf16* pa,
                                       const __bf16* ps, const __bf16* pr,
                                       const __bf16* pm, int c) {
    if (amode == 0) {
        return *(const v8bf*)(pa + c);
    } else {
        if (c < 128)      return *(const v8bf*)(ps + c);
        else if (c < 256) return *(const v8bf*)(pr + (c - 128));
        else              return *(const v8bf*)(pm + (c - 256));
    }
}

// ---------------------------------------------------------------------------
// Fused GEMM: out = [silu](A @ W + bias) [+ resid], A is M x K (K = KT*32),
// W pre-packed bf16, N = NT*16 with K*N <= 49152 (single LDS stage).
// Weights are copied to LDS ONCE per block; each wave then grid-strides over
// 16-row M-tiles, so the copy is amortized over many tiles. A-fragment loads
// are software-pipelined one k-tile ahead of the WMMA chain; with KT a
// template constant the pipeline is fully unrolled and branch-free.
// bf16-only outputs are re-tiled through a separate per-wave LDS region and
// written as coalesced 16-byte stores.
// ---------------------------------------------------------------------------
template <int NT, int KT>
__global__ void __launch_bounds__(256)
wmma_gemm(const __bf16* __restrict__ Wp, const float* __restrict__ bias, int biasN,
          int M,
          int amode, const __bf16* __restrict__ Abf, int lda,
          const __bf16* __restrict__ xb, const __bf16* __restrict__ mbuf,
          const int* __restrict__ send, const int* __restrict__ recv,
          int nodesPerB, int Ecount,
          float* __restrict__ outF, const float* __restrict__ resid, int ldoF,
          __bf16* __restrict__ outB, int ldoB,
          int storeN, int applySilu) {
    extern __shared__ __bf16 dynlds[];
    __bf16* wlds = dynlds;                 // [KT][NT][32][16] packed weights
    __bf16* clds = dynlds + WLDS_ELEMS;    // per-wave 2048-element bounce

    const int tid   = threadIdx.x;
    const int lane  = tid & 31;
    const int wave  = tid >> 5;
    const int tiles = M >> 4;

    // One-time cooperative weight copy: global -> LDS, 16B chunks.
    {
        const uint4* src  = (const uint4*)Wp;
        uint4*       dst4 = (uint4*)wlds;
        constexpr int chunks = KT * NT * 64;  // (KT*NT*512 bf16)/8 per uint4
        for (int i = tid; i < chunks; i += 256) dst4[i] = src[i];
    }
    __syncthreads();

    const int  stride  = gridDim.x * 8;
    const int  colBase = lane & 15;
    const int  kbOff   = (lane & 16) ? 16 : 0;  // contiguous k half-select (matches packed B)
    const bool bounce  = (outB != nullptr) && (outF == nullptr) && (storeN == NT * 16);

    for (int tile = blockIdx.x * 8 + wave; tile < tiles; tile += stride) {
        // Per-lane A row and source pointers.
        const int row = (tile << 4) + (lane & 15);
        const __bf16 *pa = nullptr, *ps = nullptr, *pr = nullptr, *pm = nullptr;
        if (amode == 0) {
            pa = Abf + (size_t)row * lda;
        } else {
            int b = row / Ecount;
            int e = row - b * Ecount;
            ps = xb + ((size_t)b * nodesPerB + send[e]) * H_DIM;
            pr = xb + ((size_t)b * nodesPerB + recv[e]) * H_DIM;
            pm = mbuf + (size_t)row * H_DIM;
        }

        v8f acc[NT];
        const v8f vzero = {0.f, 0.f, 0.f, 0.f, 0.f, 0.f, 0.f, 0.f};
#pragma unroll
        for (int nt = 0; nt < NT; ++nt) acc[nt] = vzero;

        // Fully unrolled, software-pipelined K chain: prefetch k-tile kt+1
        // before the WMMA chain consumes kt.
        v8bf a0 = loadA8(amode, pa, ps, pr, pm, kbOff);
        v8bf a1 = loadA8(amode, pa, ps, pr, pm, kbOff + 8);
#pragma unroll
        for (int kt = 0; kt < KT; ++kt) {
            v8bf n0 = a0, n1 = a1;
            if (kt + 1 < KT) {  // compile-time after unroll
                const int kb = ((kt + 1) << 5) + kbOff;
                n0 = loadA8(amode, pa, ps, pr, pm, kb);
                n1 = loadA8(amode, pa, ps, pr, pm, kb + 8);
            }
            v16bf a;
#pragma unroll
            for (int i = 0; i < 8; ++i) { a[i] = a0[i]; a[i + 8] = a1[i]; }
#pragma unroll
            for (int nt = 0; nt < NT; ++nt) {
                v16bf b = *(const v16bf*)&wlds[((kt * NT + nt) << 9) + (lane << 4)];
                acc[nt] = __builtin_amdgcn_wmma_f32_16x16x32_bf16(
                    /*neg_a=*/false, a, /*neg_b=*/false, b,
                    /*c_mod=*/(short)0, acc[nt],
                    /*reuse_a=*/false, /*reuse_b=*/false);
            }
            a0 = n0;
            a1 = n1;
        }

        if (bounce) {
            // Re-tile C through this wave's private LDS region, then write
            // coalesced b128 stores. NTP tiles (<=8) per sequential pass.
            constexpr int NTP   = (NT < 8) ? NT : 8;
            constexpr int COLSP = NTP * 16;
            __bf16* st = clds + (wave << 11);  // 2048 elements per wave
            const int lrow0 = (lane >> 4) << 3;
            const int lr      = lane >> 1;
            const int halfOff = (lane & 1) * (COLSP >> 1);
            const int r       = (tile << 4) + lr;
#pragma unroll
            for (int p = 0; p < NT; p += NTP) {
#pragma unroll
                for (int q = 0; q < NTP; ++q) {
                    const int nt  = p + q;
                    const int col = (nt << 4) + colBase;
                    const float bval = (col < biasN) ? bias[col] : 0.f;
                    const int lcol = (q << 4) + colBase;
#pragma unroll
                    for (int j = 0; j < 8; ++j) {
                        float v = acc[nt][j] + bval;
                        if (applySilu) v = v / (1.f + __expf(-v));
                        st[(lrow0 + j) * COLSP + lcol] = (__bf16)v;
                    }
                }
                // Two lanes per row; each copies half a row as 16B chunks.
#pragma unroll
                for (int c = 0; c < (COLSP >> 1); c += 8) {
                    v8bf v = *(const v8bf*)&st[lr * COLSP + halfOff + c];
                    *(v8bf*)&outB[(size_t)r * ldoB + (p << 4) + halfOff + c] = v;
                }
            }
        } else {
            // Scalar epilogue (fp32 out / residual / partial-column store).
            const int rbase = (tile << 4) + ((lane >> 4) << 3);
#pragma unroll
            for (int nt = 0; nt < NT; ++nt) {
                const int col = (nt << 4) + colBase;
                const float bval = (col < biasN) ? bias[col] : 0.f;
                if (col < storeN) {
#pragma unroll
                    for (int j = 0; j < 8; ++j) {
                        float v = acc[nt][j] + bval;
                        if (applySilu) v = v / (1.f + __expf(-v));
                        const int r = rbase + j;
                        float o = v;
                        if (resid) o += resid[(size_t)r * ldoF + col];
                        if (outF) outF[(size_t)r * ldoF + col] = o;
                        if (outB) outB[(size_t)r * ldoB + col] = (__bf16)o;
                    }
                }
            }
        }
    }
}

// ---------------------------------------------------------------------------
// Small helper kernels
// ---------------------------------------------------------------------------
__global__ void fill0_kernel(float* __restrict__ p, int n) {
    int t = blockIdx.x * blockDim.x + threadIdx.x;
    if (t < n) p[t] = 0.f;
}

// fp32 [rows, srcC] -> bf16 [rows, padC], zero padded
__global__ void conv_bf16_kernel(const float* __restrict__ x, __bf16* __restrict__ out,
                                 int rows, int srcC, int padC) {
    int t = blockIdx.x * blockDim.x + threadIdx.x;
    if (t >= rows * padC) return;
    int r = t / padC, c = t - r * padC;
    out[t] = (__bf16)((c < srcC) ? x[(size_t)r * srcC + c] : 0.f);
}

__global__ void count_edges_kernel(const int* __restrict__ recv, float* __restrict__ cnt, int E) {
    int t = blockIdx.x * blockDim.x + threadIdx.x;
    if (t < E) atomicAdd(&cnt[recv[t]], 1.f);
}

// agg[(b*N + recv[e])*H + h] += (float)m[(b*E + e)*H + h]
__global__ void scatter_add_kernel(const __bf16* __restrict__ m, const int* __restrict__ recv,
                                   float* __restrict__ agg) {
    size_t t = (size_t)blockIdx.x * blockDim.x + threadIdx.x;
    const size_t total = (size_t)B_DIM * E_EDGES * H_DIM;
    if (t >= total) return;
    int h = (int)(t & (H_DIM - 1));
    size_t be = t >> 7;  // /H_DIM
    int e = (int)(be % E_EDGES);
    int b = (int)(be / E_EDGES);
    atomicAdd(&agg[((size_t)b * N_NODES + recv[e]) * H_DIM + h], (float)m[t]);
}

// x = x + agg / max(cnt,1);  also refresh bf16 mirror
__global__ void combine_kernel(float* __restrict__ xf, __bf16* __restrict__ xb,
                               const float* __restrict__ agg, const float* __restrict__ cnt) {
    int t = blockIdx.x * blockDim.x + threadIdx.x;
    const int total = B_DIM * N_NODES * H_DIM;
    if (t >= total) return;
    int n = (t >> 7) % N_NODES;
    float c = cnt[n];
    c = c > 1.f ? c : 1.f;
    float v = xf[t] + agg[t] / c;
    xf[t] = v;
    xb[t] = (__bf16)v;
}

// ---------------------------------------------------------------------------
// Host-side launch helper (runtime N,K -> template NT,KT)
// ---------------------------------------------------------------------------
static inline void run_gemm(hipStream_t stream, int N,
                            const __bf16* Wp, const float* bias, int biasN, int K, int M,
                            int amode, const __bf16* Abf, int lda,
                            const __bf16* xb, const __bf16* mbuf,
                            const int* send, const int* recv,
                            float* outF, const float* resid, int ldoF,
                            __bf16* outB, int ldoB, int storeN, int applySilu) {
    int tiles  = M >> 4;
    int blocks = (tiles + 7) / 8;
    if (blocks > MAX_BLOCKS) blocks = MAX_BLOCKS;
    dim3 g(blocks), b(256);
#define GEMM_ARGS Wp, bias, biasN, M, amode, Abf, lda, xb, mbuf, \
                  send, recv, N_NODES, E_EDGES, outF, resid, ldoF, outB, ldoB, storeN, applySilu
#define DISPATCH(NTv, KTv) wmma_gemm<NTv, KTv><<<g, b, LDS_BYTES, stream>>>(GEMM_ARGS)
    if (N == 128) {
        switch (K) {
            case 64:  DISPATCH(8, 2);  break;
            case 128: DISPATCH(8, 4);  break;
            case 160: DISPATCH(8, 5);  break;
            case 256: DISPATCH(8, 8);  break;
            case 384: DISPATCH(8, 12); break;
            default: break;
        }
    } else if (N == 256 && K == 128) {
        DISPATCH(16, 4);
    } else if (N == 32 && K == 128) {
        DISPATCH(2, 4);
    }
#undef DISPATCH
#undef GEMM_ARGS
}

// ---------------------------------------------------------------------------
// kernel_launch
// ---------------------------------------------------------------------------
extern "C" void kernel_launch(void* const* d_in, const int* in_sizes, int n_in,
                              void* d_out, int out_size, void* d_ws, size_t ws_size,
                              hipStream_t stream) {
    (void)in_sizes; (void)n_in; (void)out_size; (void)ws_size;
    const int Bv = B_DIM, Nv = N_NODES, Ev = E_EDGES;
    const int INv = 60, Hv = 128, EFv = 154, OUTv = 30;
    const int rowsN = Bv * Nv;   // 10000
    const int rowsE = Bv * Ev;   // 320000

    // ---- inputs (insertion order of setup_inputs dict) ----
    const float* x_in      = (const float*)d_in[0];
    const float* edge_attr = (const float*)d_in[1];
    const int*   send      = (const int*)d_in[2];
    const int*   recv      = (const int*)d_in[3];
    int idx = 4;
    struct Layer { const float *mw1,*mb1,*mw2,*mb2,*uw1,*ub1,*uw2,*ub2,*rw,*rb; };
    Layer L[4];
    for (int l = 0; l < 4; ++l) {
        L[l].mw1 = (const float*)d_in[idx++]; L[l].mb1 = (const float*)d_in[idx++];
        L[l].mw2 = (const float*)d_in[idx++]; L[l].mb2 = (const float*)d_in[idx++];
        L[l].uw1 = (const float*)d_in[idx++]; L[l].ub1 = (const float*)d_in[idx++];
        L[l].uw2 = (const float*)d_in[idx++]; L[l].ub2 = (const float*)d_in[idx++];
        if (l == 0) { L[l].rw = (const float*)d_in[idx++]; L[l].rb = (const float*)d_in[idx++]; }
        else        { L[l].rw = nullptr; L[l].rb = nullptr; }
    }
    const float* ow1 = (const float*)d_in[idx++]; const float* ob1 = (const float*)d_in[idx++];
    const float* ow2 = (const float*)d_in[idx++]; const float* ob2 = (const float*)d_in[idx++];
    const float* ow3 = (const float*)d_in[idx++]; const float* ob3 = (const float*)d_in[idx++];

    // ---- workspace carve ----
    char* wsp = (char*)d_ws;
    auto carve = [&](size_t bytes) -> void* {
        void* r = (void*)wsp;
        wsp += (bytes + 255) & ~(size_t)255;
        return r;
    };

    // Packed weights (bf16, WMMA B layout).
    struct PackJob { const float* src; int sK, sN, pK, pN; __bf16* dst; };
    PackJob jobs[20];
    int nj = 0;
    __bf16 *Wp_mw1[4], *Wp_mw2[4], *Wp_uw1[4], *Wp_uw2[4], *Wp_rw, *Wp_o1, *Wp_o2, *Wp_o3;
    for (int l = 0; l < 4; ++l) {
        int mK  = (l == 0) ? EFv : 3 * Hv;   // 154 or 384
        int mKp = (l == 0) ? 160 : 384;
        Wp_mw1[l] = (__bf16*)carve((size_t)mKp * Hv * 2);
        jobs[nj++] = {L[l].mw1, mK, Hv, mKp, Hv, Wp_mw1[l]};
        Wp_mw2[l] = (__bf16*)carve((size_t)Hv * Hv * 2);
        jobs[nj++] = {L[l].mw2, Hv, Hv, Hv, Hv, Wp_mw2[l]};
        Wp_uw1[l] = (__bf16*)carve((size_t)Hv * 2 * Hv * 2);
        jobs[nj++] = {L[l].uw1, Hv, 2 * Hv, Hv, 2 * Hv, Wp_uw1[l]};
        Wp_uw2[l] = (__bf16*)carve((size_t)2 * Hv * Hv * 2);
        jobs[nj++] = {L[l].uw2, 2 * Hv, Hv, 2 * Hv, Hv, Wp_uw2[l]};
    }
    Wp_rw = (__bf16*)carve((size_t)64 * Hv * 2);
    jobs[nj++] = {L[0].rw, INv, Hv, 64, Hv, Wp_rw};
    Wp_o1 = (__bf16*)carve((size_t)Hv * Hv * 2); jobs[nj++] = {ow1, Hv, Hv, Hv, Hv, Wp_o1};
    Wp_o2 = (__bf16*)carve((size_t)Hv * Hv * 2); jobs[nj++] = {ow2, Hv, Hv, Hv, Hv, Wp_o2};
    Wp_o3 = (__bf16*)carve((size_t)Hv * 32 * 2); jobs[nj++] = {ow3, Hv, OUTv, Hv, 32, Wp_o3};

    __bf16* edge_b = (__bf16*)carve((size_t)rowsE * 160 * 2);  // padded bf16 edge_attr
    __bf16* xin_b  = (__bf16*)carve((size_t)rowsN * 64 * 2);
    float*  xf     = (float*) carve((size_t)rowsN * Hv * 4);
    __bf16* xbf    = (__bf16*)carve((size_t)rowsN * Hv * 2);
    float*  agg    = (float*) carve((size_t)rowsN * Hv * 4);
    float*  cnt    = (float*) carve((size_t)Nv * 4);
    __bf16* m1     = (__bf16*)carve((size_t)rowsE * Hv * 2);
    __bf16* mcur   = (__bf16*)carve((size_t)rowsE * Hv * 2);
    __bf16* u1     = (__bf16*)carve((size_t)rowsN * 2 * Hv * 2);
    __bf16* h1     = (__bf16*)carve((size_t)rowsN * Hv * 2);
    __bf16* h2     = (__bf16*)carve((size_t)rowsN * Hv * 2);

    // ---- weight packing (cheap; runs every call -> deterministic) ----
    for (int j = 0; j < nj; ++j) {
        int threads = (jobs[j].pK >> 5) * (jobs[j].pN >> 4) * 32;
        pack_w_kernel<<<(threads + 255) / 256, 256, 0, stream>>>(
            jobs[j].src, jobs[j].dst, jobs[j].sK, jobs[j].sN, jobs[j].pK, jobs[j].pN);
    }

    // ---- input conversions to bf16 (coalesced, vector-friendly) ----
    conv_bf16_kernel<<<(rowsN * 64 + 255) / 256, 256, 0, stream>>>(x_in, xin_b, rowsN, INv, 64);
    {
        long long tot = (long long)rowsE * 160;
        conv_bf16_kernel<<<(unsigned)((tot + 255) / 256), 256, 0, stream>>>(
            edge_attr, edge_b, rowsE, EFv, 160);
    }

    // ---- edge counts (independent of batch) ----
    fill0_kernel<<<(Nv + 255) / 256, 256, 0, stream>>>(cnt, Nv);
    count_edges_kernel<<<(Ev + 255) / 256, 256, 0, stream>>>(recv, cnt, Ev);

    // ---- GNN layers ----
    for (int l = 0; l < 4; ++l) {
        // m1 = silu(concat/edge_attr @ mw1 + mb1)
        if (l == 0) {
            run_gemm(stream, Hv, Wp_mw1[0], L[0].mb1, Hv, /*K=*/160, rowsE,
                     /*amode=*/0, edge_b, 160, nullptr, nullptr, nullptr, nullptr,
                     nullptr, nullptr, 0, m1, Hv, Hv, /*silu=*/1);
        } else {
            run_gemm(stream, Hv, Wp_mw1[l], L[l].mb1, Hv, /*K=*/384, rowsE,
                     /*amode=*/2, nullptr, 0, xbf, mcur, send, recv,
                     nullptr, nullptr, 0, m1, Hv, Hv, /*silu=*/1);
        }
        // mcur = silu(m1 @ mw2 + mb2)
        run_gemm(stream, Hv, Wp_mw2[l], L[l].mb2, Hv, /*K=*/Hv, rowsE,
                 /*amode=*/0, m1, Hv, nullptr, nullptr, nullptr, nullptr,
                 nullptr, nullptr, 0, mcur, Hv, Hv, /*silu=*/1);

        // agg = scatter_sum(mcur)
        fill0_kernel<<<(rowsN * Hv + 255) / 256, 256, 0, stream>>>(agg, rowsN * Hv);
        {
            size_t total = (size_t)rowsE * Hv;
            scatter_add_kernel<<<(unsigned)((total + 255) / 256), 256, 0, stream>>>(mcur, recv, agg);
        }

        // layer-0 residual projection: xf = x_in @ rw + rb
        if (l == 0) {
            run_gemm(stream, Hv, Wp_rw, L[0].rb, Hv, /*K=*/64, rowsN,
                     /*amode=*/0, xin_b, 64, nullptr, nullptr, nullptr, nullptr,
                     xf, nullptr, Hv, nullptr, 0, Hv, /*silu=*/0);
        }

        // x = res + agg / max(cnt,1)   (in place on xf, refresh xbf)
        combine_kernel<<<(rowsN * Hv + 255) / 256, 256, 0, stream>>>(xf, xbf, agg, cnt);

        // u1 = silu(x @ uw1 + ub1)   (N=256)
        run_gemm(stream, 2 * Hv, Wp_uw1[l], L[l].ub1, 2 * Hv, /*K=*/Hv, rowsN,
                 /*amode=*/0, xbf, Hv, nullptr, nullptr, nullptr, nullptr,
                 nullptr, nullptr, 0, u1, 2 * Hv, 2 * Hv, /*silu=*/1);

        // x = x + (u1 @ uw2 + ub2)   (residual fused; refresh xbf)
        run_gemm(stream, Hv, Wp_uw2[l], L[l].ub2, Hv, /*K=*/2 * Hv, rowsN,
                 /*amode=*/0, u1, 2 * Hv, nullptr, nullptr, nullptr, nullptr,
                 xf, xf, Hv, xbf, Hv, Hv, /*silu=*/0);
    }

    // ---- output head ----
    run_gemm(stream, Hv, Wp_o1, ob1, Hv, Hv, rowsN,
             0, xbf, Hv, nullptr, nullptr, nullptr, nullptr,
             nullptr, nullptr, 0, h1, Hv, Hv, 1);
    run_gemm(stream, Hv, Wp_o2, ob2, Hv, Hv, rowsN,
             0, h1, Hv, nullptr, nullptr, nullptr, nullptr,
             nullptr, nullptr, 0, h2, Hv, Hv, 1);
    run_gemm(stream, 32, Wp_o3, ob3, OUTv, Hv, rowsN,
             0, h2, Hv, nullptr, nullptr, nullptr, nullptr,
             (float*)d_out, nullptr, OUTv, nullptr, 0, OUTv, 0);
}
